// SpMiddleFHDHRZ_197568495914
// MI455X (gfx1250) — compile-verified, matrix operand order checked by reading, not run on hardware
//
#include <hip/hip_runtime.h>

typedef __attribute__((ext_vector_type(16))) _Float16 v16h;
typedef __attribute__((ext_vector_type(8)))  float    v8f;

// ---------------------------------------------------------------------------
// Utility
// ---------------------------------------------------------------------------
static inline int cdiv_i(long a, long b) { return (int)((a + b - 1) / b); }

// ---------------------------------------------------------------------------
// Zero-fill (graph-capture-safe replacement for hipMemsetAsync)
// ---------------------------------------------------------------------------
__global__ void zero_f16_kernel(_Float16* __restrict__ p, long n) {
  long i = ((long)blockIdx.x * blockDim.x + threadIdx.x) * 8;
  if (i + 8 <= n) {
    float4 z; z.x = 0.f; z.y = 0.f; z.z = 0.f; z.w = 0.f;
    *(float4*)(p + i) = z;
  } else {
    for (long j = i; j < n; ++j) p[j] = (_Float16)0.f;
  }
}

// ---------------------------------------------------------------------------
// Scatter voxel features into dense channels-last f16 grid + occupancy mask
// ---------------------------------------------------------------------------
__global__ void scatter_kernel(const float* __restrict__ vf,
                               const int* __restrict__ coors,
                               _Float16* __restrict__ x,
                               _Float16* __restrict__ mask,
                               int nvox, int C, int D, int H, int W) {
  int v = blockIdx.x;
  int c = threadIdx.x;
  if (v >= nvox || c >= C) return;
  int b = coors[v * 4 + 0];
  int d = coors[v * 4 + 1];
  int h = coors[v * 4 + 2];
  int w = coors[v * 4 + 3];
  long pos = (((long)b * D + d) * H + h) * (long)W + w;
  x[pos * C + c] = (_Float16)vf[(long)v * C + c];
  if (c == 0) mask[pos] = (_Float16)1.f;
}

// ---------------------------------------------------------------------------
// Repack weights: f32 [cout][cin][kd][kh][kw] -> f16 [tap][cin][cout]
// ---------------------------------------------------------------------------
__global__ void wpack_kernel(const float* __restrict__ Wsrc,
                             _Float16* __restrict__ Wdst,
                             int cout, int cin, int KD, int KH, int KW) {
  long n = (long)cout * cin * KD * KH * KW;
  long i = (long)blockIdx.x * blockDim.x + threadIdx.x;
  if (i >= n) return;
  int co = (int)(i % cout);
  long t = i / cout;
  int ci = (int)(t % cin);
  int tap = (int)(t / cin);
  int kw = tap % KW;
  int t2 = tap / KW;
  int kh = t2 % KH;
  int kd = t2 / KH;
  float val = Wsrc[((((long)co * cin + ci) * KD + kd) * KH + kh) * KW + kw];
  Wdst[i] = (_Float16)val;
}

// ---------------------------------------------------------------------------
// Fold BN into per-channel scale/shift
// ---------------------------------------------------------------------------
__global__ void bnprep_kernel(const float* __restrict__ bn,
                              float* __restrict__ scale,
                              float* __restrict__ shift, int cout) {
  int c = blockIdx.x * blockDim.x + threadIdx.x;
  if (c >= cout) return;
  float g = bn[c];
  float b = bn[cout + c];
  float m = bn[2 * cout + c];
  float v = bn[3 * cout + c];
  float s = g * rsqrtf(v + 1e-3f);
  scale[c] = s;
  shift[c] = b - m * s;
}

// ---------------------------------------------------------------------------
// Mask down-sample for strided ('sp') layers: any() over the conv window
// ---------------------------------------------------------------------------
__global__ void maskpool_kernel(const _Float16* __restrict__ min_,
                                _Float16* __restrict__ mout,
                                int ID, int IH, int IW,
                                int OD, int OH, int OW,
                                int KD, int KH, int KW,
                                int SD, int SH, int SW,
                                int PD, int PH, int PW, int Mtotal) {
  int row = blockIdx.x * blockDim.x + threadIdx.x;
  if (row >= Mtotal) return;
  int t = row;
  int ow = t % OW; t /= OW;
  int oh = t % OH; t /= OH;
  int od = t % OD;
  int b = t / OD;
  float any = 0.f;
  for (int kd = 0; kd < KD; ++kd) {
    int id = od * SD - PD + kd;
    if (id < 0 || id >= ID) continue;
    for (int kh = 0; kh < KH; ++kh) {
      int ih = oh * SH - PH + kh;
      if (ih < 0 || ih >= IH) continue;
      for (int kw = 0; kw < KW; ++kw) {
        int iw = ow * SW - PW + kw;
        if (iw < 0 || iw >= IW) continue;
        if ((float)min_[(((long)b * ID + id) * IH + ih) * (long)IW + iw] > 0.f)
          any = 1.f;
      }
    }
  }
  mout[row] = (_Float16)any;
}

// ---------------------------------------------------------------------------
// Implicit-GEMM conv3d with WMMA f32_16x16x32_f16, fused BN+ReLU+mask.
//   x: [B][ID][IH][IW][CIN]  (f16, channels-last)
//   w: [taps][CIN][Cout]     (f16)
//   y: [B][OD][OH][OW][Cout] (f16)
// - templated on CIN + kernel dims + NT (N-tiles per wave): fully unrolled
//   flat (tap,kc) loop, host-precomputed tap offsets in kernarg -> no div/mod
// - per-thread 27-bit tap validity mask computed once
// - double-buffered LDS + 2-deep register staging: ONE barrier per K-step
// - NT=2 for Cout>=32: one A-fragment feeds two v_wmma (16x32 per wave),
//   halving barriers + A traffic per matrix op
// Block = 256 threads = 8 waves -> 128 output positions x 16*NT channels.
// ---------------------------------------------------------------------------
struct ConvArgs {
  const _Float16* x;
  const _Float16* w;
  const float* scale;
  const float* shift;
  const _Float16* mask;  // at OUTPUT resolution, flat [Mtotal]
  _Float16* y;
  int Cout;
  int ID, IH, IW, OD, OH, OW;
  int SD, SH, SW, PD, PH, PW;
  int Mtotal;
  int tapOff[27];        // ((kd*IH + kh)*IW + kw) * CIN  (element offsets)
};

template <int CIN, int KD, int KH, int KW, int NT>
__global__ __launch_bounds__(256) void conv3d_wmma_kernel(ConvArgs p) {
  constexpr int KC = (CIN + 31) / 32;  // 32-channel K chunks
  constexpr int TAPS = KD * KH * KW;
  constexpr int NITER = TAPS * KC;
  constexpr int ABUF = 128 * 32;       // halves per A buffer
  constexpr int BBUF = 16 * NT * 32;   // halves per B buffer

  __shared__ __align__(32) _Float16 lA[2 * ABUF];   // ping-pong A tiles
  __shared__ __align__(32) _Float16 lBt[2 * BBUF];  // ping-pong B tiles (transposed)
  __shared__ float lMask[128];

  const int tid = threadIdx.x;
  const int lane = tid & 31;
  const int wv = tid >> 5;             // 0..7 -> M sub-tile
  const int Mbase = blockIdx.x * 128;
  const int nbase = blockIdx.y * (16 * NT);

  // ---- per-thread A-row descriptor -> registers ----
  const int arow = tid >> 1;           // 0..127
  const int aseg = tid & 1;            // 16-half segment within 32-chunk
  int ab = -1, ad = 0, ah = 0, aw = 0;
  {
    int row = Mbase + arow;
    if (row < p.Mtotal) {
      int t = row;
      int ow_ = t % p.OW; t /= p.OW;
      int oh_ = t % p.OH; t /= p.OH;
      int od_ = t % p.OD;
      ab = t / p.OD;
      ad = od_ * p.SD - p.PD;
      ah = oh_ * p.SH - p.PH;
      aw = ow_ * p.SW - p.PW;
    }
  }
  if (tid < 128) {
    int row = Mbase + tid;
    lMask[tid] = (row < p.Mtotal) ? (float)p.mask[row] : 0.f;
  }

  // Base pointer at tap (0,0,0); only dereferenced for valid taps.
  const _Float16* abase = p.x +
      ((((long)ab * p.ID + ad) * p.IH + ah) * (long)p.IW + aw) * CIN + aseg * 16;

  // 27-bit per-thread tap validity mask, computed once.
  unsigned vmask = 0;
  {
    int tap = 0;
#pragma unroll
    for (int kd = 0; kd < KD; ++kd)
#pragma unroll
      for (int kh = 0; kh < KH; ++kh)
#pragma unroll
        for (int kw = 0; kw < KW; ++kw, ++tap) {
          int id = ad + kd, ih = ah + kh, iw = aw + kw;
          bool ok = (ab >= 0) & (id >= 0) & (id < p.ID) & (ih >= 0) &
                    (ih < p.IH) & (iw >= 0) & (iw < p.IW);
          vmask |= ((unsigned)ok) << tap;
        }
  }

  // ---- per-thread B descriptor: k = tid&31, NT n-pairs per thread ----
  const int bk = tid & 31;
  const int bng = tid >> 5;  // 0..7 -> n pair within each 16-wide tile
  const _Float16* wbase = p.w + (long)bk * p.Cout + nbase + bng * 2;

  float4 ra0, ra1;          // staged A registers (32 bytes)
  _Float16 rb[NT][2];       // staged B registers

  auto stage = [&](int it) {
    const int tap = it / KC;   // constant after full unroll
    const int kc = it % KC;
    // A
    ra0.x = ra0.y = ra0.z = ra0.w = 0.f;
    ra1 = ra0;
    bool segok = (kc * 32 + aseg * 16) < CIN;
    if (((vmask >> tap) & 1u) && segok) {
      const float4* src = (const float4*)(abase + (long)p.tapOff[tap] + kc * 32);
      ra0 = src[0];
      ra1 = src[1];
    }
    // B
#pragma unroll
    for (int j = 0; j < NT; ++j) {
      rb[j][0] = (_Float16)0.f;
      rb[j][1] = (_Float16)0.f;
    }
    if (kc * 32 + bk < CIN) {
      const _Float16* s = wbase + (long)(tap * CIN + kc * 32) * p.Cout;
#pragma unroll
      for (int j = 0; j < NT; ++j) {
        rb[j][0] = s[j * 16 + 0];
        rb[j][1] = s[j * 16 + 1];
      }
    }
  };

  const int aoff = arow * 32 + aseg * 16;
  auto commit = [&](int buf) {
    _Float16* A = lA + buf * ABUF;
    _Float16* Bt = lBt + buf * BBUF;
    *(float4*)(A + aoff) = ra0;
    *(float4*)(A + aoff + 8) = ra1;
#pragma unroll
    for (int j = 0; j < NT; ++j) {
      Bt[(j * 16 + bng * 2 + 0) * 32 + bk] = rb[j][0];
      Bt[(j * 16 + bng * 2 + 1) * 32 + bk] = rb[j][1];
    }
  };

  const int afoff = (wv * 16 + (lane & 15)) * 32 + (lane >> 4) * 16;
  const int bfoff = (lane & 15) * 32 + (lane >> 4) * 16;

  v8f acc[NT] = {};
  stage(0);
  commit(0);
  if (NITER > 1) stage(1);   // in flight across the first barrier
  __syncthreads();

#pragma unroll
  for (int it = 0; it < NITER; ++it) {
    if (it + 1 < NITER) commit((it + 1) & 1);  // waits on loads issued 1 iter ago
    if (it + 2 < NITER) stage(it + 2);         // issue loads 2 iters ahead
    const _Float16* A = lA + (it & 1) * ABUF;
    const _Float16* Bt = lBt + (it & 1) * BBUF;
    v16h afrag = *(const v16h*)(A + afoff);
#pragma unroll
    for (int j = 0; j < NT; ++j) {
      v16h bfrag = *(const v16h*)(Bt + bfoff + j * (16 * 32));
      acc[j] = __builtin_amdgcn_wmma_f32_16x16x32_f16(
          /*neg_a=*/false, afrag, /*neg_b=*/false, bfrag,
          /*c_mod=*/(short)0, acc[j], /*reuse_a=*/false, /*reuse_b=*/false);
    }
    __syncthreads();  // one barrier per K-step
  }

  // ---- epilogue: BN scale/shift + ReLU + mask, store f16 channels-last ----
  int rloc0 = wv * 16 + ((lane >> 4) << 3);  // C/D layout: M = vgpr + 8*(lane/16)
#pragma unroll
  for (int j = 0; j < NT; ++j) {
    int col = nbase + j * 16 + (lane & 15);
    float sc = p.scale[col];
    float sh = p.shift[col];
#pragma unroll
    for (int i = 0; i < 8; ++i) {
      int rloc = rloc0 + i;
      int row = Mbase + rloc;
      if (row < p.Mtotal) {
        float v = acc[j][i] * sc + sh;
        v = fmaxf(v, 0.f) * lMask[rloc];
        p.y[(long)row * p.Cout + col] = (_Float16)v;
      }
    }
  }
}

// ---------------------------------------------------------------------------
// Channels-last f16 -> NCDHW-flattened f32 output
// ---------------------------------------------------------------------------
__global__ void finalize_kernel(const _Float16* __restrict__ x,
                                float* __restrict__ out,
                                int Bsz, int C, int D, int H, int W) {
  int i = blockIdx.x * blockDim.x + threadIdx.x;
  int total = Bsz * C * D * H * W;
  if (i >= total) return;
  int t = i;
  int w = t % W; t /= W;
  int h = t % H; t /= H;
  int d = t % D; t /= D;
  int c = t % C;
  int b = t / C;
  out[i] = (float)x[((((long)b * D + d) * H + h) * (long)W + w) * C + c];
}

// ---------------------------------------------------------------------------
// Host-side orchestration
// ---------------------------------------------------------------------------
extern "C" void kernel_launch(void* const* d_in, const int* in_sizes, int n_in,
                              void* d_out, int out_size, void* d_ws, size_t ws_size,
                              hipStream_t stream) {
  (void)n_in; (void)ws_size;

  static const int S_KIND[15] = {0,0,1,0,0,1,0,0,1,0,0,1,0,0,1};  // 1 = strided 'sp'
  static const int S_CIN[15]  = {128,16,16,32,32,32,64,64,64,64,64,64,64,64,64};
  static const int S_COUT[15] = {16,16,32,32,32,64,64,64,64,64,64,64,64,64,64};
  static const int S_K[15][3] = {{3,3,3},{3,3,3},{3,3,3},{3,3,3},{3,3,3},{3,3,3},
                                 {3,3,3},{3,3,3},{3,3,3},{3,3,3},{3,3,3},{3,1,1},
                                 {3,3,3},{3,3,3},{3,1,1}};
  static const int S_S[15][3] = {{1,1,1},{1,1,1},{2,2,2},{1,1,1},{1,1,1},{2,2,2},
                                 {1,1,1},{1,1,1},{2,2,2},{1,1,1},{1,1,1},{2,1,1},
                                 {1,1,1},{1,1,1},{2,1,1}};
  static const int S_P[15][3] = {{1,1,1},{1,1,1},{1,1,1},{1,1,1},{1,1,1},{1,1,1},
                                 {1,1,1},{1,1,1},{1,1,1},{1,1,1},{1,1,1},{0,0,0},
                                 {1,1,1},{1,1,1},{0,0,0}};

  const float* vf = (const float*)d_in[0];
  const int* coors = (const int*)d_in[1];
  const int C0 = 128;
  int nvox = in_sizes[0] / C0;

  const int Bs = 2, D0 = 49, H0 = 96, Wd0 = 96;
  const size_t P0 = (size_t)Bs * D0 * H0 * Wd0;

  // ---- carve workspace ----
  char* wsb = (char*)d_ws;
  size_t off = 0;
  auto carve = [&](size_t bytes) -> char* {
    off = (off + 255) & ~(size_t)255;
    char* ptr = wsb + off;
    off += bytes;
    return ptr;
  };
  _Float16* xA = (_Float16*)carve(P0 * 128 * sizeof(_Float16));  // big ping
  _Float16* xB = (_Float16*)carve(P0 * 16 * sizeof(_Float16));   // pong
  _Float16* mA = (_Float16*)carve(P0 * sizeof(_Float16));
  _Float16* mB = (_Float16*)carve(P0 * sizeof(_Float16));

  size_t woff[15], wtot = 0;
  for (int l = 0; l < 15; ++l) {
    woff[l] = wtot;
    wtot += (size_t)S_K[l][0] * S_K[l][1] * S_K[l][2] * S_CIN[l] * S_COUT[l];
  }
  _Float16* wbuf = (_Float16*)carve(wtot * sizeof(_Float16));
  float* scbuf = (float*)carve(15 * 64 * sizeof(float));
  float* shbuf = (float*)carve(15 * 64 * sizeof(float));

  // ---- zero the dense grid + mask, scatter voxels ----
  {
    long nx = (long)P0 * 128;
    zero_f16_kernel<<<cdiv_i(cdiv_i(nx, 8), 256), 256, 0, stream>>>(xA, nx);
    long nm = (long)P0;
    zero_f16_kernel<<<cdiv_i(cdiv_i(nm, 8), 256), 256, 0, stream>>>(mA, nm);
  }
  scatter_kernel<<<nvox, C0, 0, stream>>>(vf, coors, xA, mA, nvox, C0, D0, H0, Wd0);

  // ---- layer loop ----
  int dims[3] = {D0, H0, Wd0};
  _Float16* curX = xA; _Float16* nxtX = xB;
  _Float16* curM = mA; _Float16* nxtM = mB;

  for (int l = 0; l < 15; ++l) {
    const int cin = S_CIN[l], cout = S_COUT[l];
    const int KD = S_K[l][0], KH = S_K[l][1], KW = S_K[l][2];
    const int SD = S_S[l][0], SH = S_S[l][1], SW = S_S[l][2];
    const int PD = S_P[l][0], PH = S_P[l][1], PW = S_P[l][2];

    int od = dims[0], oh = dims[1], ow = dims[2];
    if (S_KIND[l] == 1) {
      od = (dims[0] + 2 * PD - KD) / SD + 1;
      oh = (dims[1] + 2 * PH - KH) / SH + 1;
      ow = (dims[2] + 2 * PW - KW) / SW + 1;
    }
    const long Mout = (long)Bs * od * oh * ow;

    // weight repack + BN fold (tiny)
    {
      long nw = (long)KD * KH * KW * cin * cout;
      wpack_kernel<<<cdiv_i(nw, 256), 256, 0, stream>>>(
          (const float*)d_in[3 + l], wbuf + woff[l], cout, cin, KD, KH, KW);
      bnprep_kernel<<<1, 64, 0, stream>>>(
          (const float*)d_in[18 + l], scbuf + l * 64, shbuf + l * 64, cout);
    }

    if (S_KIND[l] == 1) {
      maskpool_kernel<<<cdiv_i(Mout, 256), 256, 0, stream>>>(
          curM, nxtM, dims[0], dims[1], dims[2], od, oh, ow,
          KD, KH, KW, SD, SH, SW, PD, PH, PW, (int)Mout);
      _Float16* tm = curM; curM = nxtM; nxtM = tm;
    }

    ConvArgs a;
    a.x = curX; a.w = wbuf + woff[l];
    a.scale = scbuf + l * 64; a.shift = shbuf + l * 64;
    a.mask = curM; a.y = nxtX;
    a.Cout = cout;
    a.ID = dims[0]; a.IH = dims[1]; a.IW = dims[2];
    a.OD = od; a.OH = oh; a.OW = ow;
    a.SD = SD; a.SH = SH; a.SW = SW;
    a.PD = PD; a.PH = PH; a.PW = PW;
    a.Mtotal = (int)Mout;
    // host-side tap offset table: ((kd*IH + kh)*IW + kw) * CIN
    {
      int tap = 0;
      for (int kd = 0; kd < KD; ++kd)
        for (int kh = 0; kh < KH; ++kh)
          for (int kw = 0; kw < KW; ++kw, ++tap)
            a.tapOff[tap] = ((kd * dims[1] + kh) * dims[2] + kw) * cin;
      for (; tap < 27; ++tap) a.tapOff[tap] = 0;
    }

    const int nt = (cout >= 32) ? 2 : 1;
    dim3 grid((unsigned)cdiv_i(Mout, 128), (unsigned)(cout / (16 * nt)), 1);
    if (cin == 128) {
      conv3d_wmma_kernel<128, 3, 3, 3, 1><<<grid, 256, 0, stream>>>(a);
    } else if (cin == 16 && cout == 16) {
      conv3d_wmma_kernel<16, 3, 3, 3, 1><<<grid, 256, 0, stream>>>(a);
    } else if (cin == 16) {
      conv3d_wmma_kernel<16, 3, 3, 3, 2><<<grid, 256, 0, stream>>>(a);
    } else if (cin == 32) {
      conv3d_wmma_kernel<32, 3, 3, 3, 2><<<grid, 256, 0, stream>>>(a);
    } else if (KH == 1) {
      conv3d_wmma_kernel<64, 3, 1, 1, 2><<<grid, 256, 0, stream>>>(a);
    } else {
      conv3d_wmma_kernel<64, 3, 3, 3, 2><<<grid, 256, 0, stream>>>(a);
    }

    _Float16* tx = curX; curX = nxtX; nxtX = tx;
    dims[0] = od; dims[1] = oh; dims[2] = ow;
  }

  // ---- final transpose to f32 NCDHW-flattened output ----
  finalize_kernel<<<cdiv_i(out_size, 256), 256, 0, stream>>>(
      curX, (float*)d_out, Bs, 64, dims[0], dims[1], dims[2]);
}
